// NodeEmbUpd_25563645346107
// MI455X (gfx1250) — matrix operand order repaired and checked by reading, not compiled, not run on hardware
//
#include <hip/hip_runtime.h>
#include <hip/hip_bf16.h>

// ---------------------------------------------------------------------------
// Bidirectional GNN (message GEMM + scatter-add + GRU cell), CDNA5 / gfx1250.
//
// Roofline (per layer, per direction):
//   message GEMM:  E*128*128*2 = 26 GFLOP  -> v_wmma_f32_16x16x32_bf16
//   edge gather:   E*2*64 elems, bf16 = 205 MB  (dominant; fp32 would be 410 MB)
//   scatter-add:   E*128 f32 atomics = 410 MB
//   GRU GEMMs:     ~3.7 GFLOP -> WMMA as well
// Memory-bound on gather/scatter -> bf16 activations halve the dominant
// traffic; f32 WMMA accumulation keeps the GRU recurrences stable.
// ---------------------------------------------------------------------------

#define N_NODES 50000
#define N_EDGES 800000
#define NDIM    128
#define HD      64
#define NLAYERS 2

typedef __bf16 bf16_t;
typedef __bf16 v16bf __attribute__((ext_vector_type(16)));
typedef __bf16 v8bf  __attribute__((ext_vector_type(8)));
typedef float  v8f   __attribute__((ext_vector_type(8)));

static __device__ __forceinline__ v8f wmma_bf16(v16bf a, v16bf b, v8f c) {
  return __builtin_amdgcn_wmma_f32_16x16x32_bf16(
      /*neg_a=*/false, a, /*neg_b=*/false, b,
      /*c_mod=*/(short)0, c, /*reuse_a=*/false, /*reuse_b=*/false);
}

// ---------------------------------------------------------------------- prep
__global__ void prep_nodes_kernel(const float* __restrict__ h,
                                  float* __restrict__ hsF, float* __restrict__ hsB,
                                  bf16_t* __restrict__ hsFb, bf16_t* __restrict__ hsBb) {
  int i = blockIdx.x * blockDim.x + threadIdx.x;
  if (i >= N_NODES * HD) return;
  int n = i / HD, c = i % HD;
  float f = h[(long long)n * NDIM + c];
  float b = h[(long long)n * NDIM + HD + c];
  hsF[i] = f; hsB[i] = b;
  hsFb[i] = (bf16_t)f; hsBb[i] = (bf16_t)b;
}

__global__ void f32_to_bf16_kernel(const float* __restrict__ src,
                                   bf16_t* __restrict__ dst, int n) {
  int i = blockIdx.x * blockDim.x + threadIdx.x;
  if (i < n) dst[i] = (bf16_t)src[i];
}

__global__ void zero_f32_kernel(float* __restrict__ p, int n) {
  int i = blockIdx.x * blockDim.x + threadIdx.x;
  if (i < n) p[i] = 0.0f;
}

// -------------------------------------------------------------- message pass
// One wave = one 16-edge tile.  m[e] = concat(hs[gsrc[e]], hs[gdst[e]]) @ Wm^T + bm
// scatter-added into aggr[gdst[e]].
__global__ __launch_bounds__(256)
void msg_kernel(const bf16_t* __restrict__ hsb,        // [N, HD] bf16
                const long long* __restrict__ gsrc,    // first gather role
                const long long* __restrict__ gdst,    // second gather role + scatter target
                const bf16_t* __restrict__ wm,         // [NDIM, NDIM] bf16 (row n = output col)
                const float*  __restrict__ bm,         // [NDIM]
                float* __restrict__ aggr)              // [N, NDIM] f32
{
  __shared__ bf16_t ldsW[NDIM * NDIM];                 // 32 KB
  {
    const unsigned* s = (const unsigned*)wm;
    unsigned* d = (unsigned*)ldsW;
    for (int i = threadIdx.x; i < NDIM * NDIM / 2; i += 256) d[i] = s[i];
  }
  __syncthreads();

  const int lane  = threadIdx.x & 31;
  const int wave  = threadIdx.x >> 5;
  const int tile  = blockIdx.x * 8 + wave;
  const long long ebase = (long long)tile * 16;
  if (ebase >= N_EDGES) return;          // wave-uniform: EXEC stays all-1s

  const int mcol  = lane & 15;           // A: row  /  B: column
  const int khalf = lane >> 4;           // K-half select (contiguous k = 16*khalf + s)
  const int lo    = khalf * 16;

  const long long e  = ebase + mcol;
  const long long ns = gsrc[e];
  const long long nd = gdst[e];
  const bf16_t* rs = hsb + ns * HD;
  const bf16_t* rd = hsb + nd * HD;

  v8f acc[8];
#pragma unroll
  for (int t = 0; t < 8; ++t)
#pragma unroll
    for (int r = 0; r < 8; ++r) acc[t][r] = 0.0f;

#pragma unroll
  for (int s = 0; s < 4; ++s) {          // K = 128 in 4 steps of 32
    const bf16_t* rp = (s < 2) ? rs : rd;
    const int fb = (s & 1) * 32;
    v8bf a0 = *(const v8bf*)(rp + fb + lo);
    v8bf a1 = *(const v8bf*)(rp + fb + lo + 8);
    v16bf A;
#pragma unroll
    for (int i = 0; i < 8; ++i) { A[i] = a0[i]; A[8 + i] = a1[i]; }
    const int k0 = s * 32 + khalf * 16;
#pragma unroll
    for (int t = 0; t < 8; ++t) {        // 8 output column tiles (128 cols)
      const int n = t * 16 + mcol;
      v16bf B = *(const v16bf*)(ldsW + n * NDIM + k0);
      acc[t] = wmma_bf16(A, B, acc[t]);
    }
  }

  // epilogue: +bias, scatter-add into aggr[dst]
#pragma unroll
  for (int t = 0; t < 8; ++t) {
    const int n = t * 16 + mcol;
    const float bias = bm[n];
#pragma unroll
    for (int r = 0; r < 8; ++r) {
      const int m = r + khalf * 8;       // C-layout: vgpr r -> row r / r+8
      const long long d = gdst[ebase + m];
      unsafeAtomicAdd(&aggr[d * NDIM + n], acc[t][r] + bias);
    }
  }
}

// ---------------------------------------------------------------- GRU update
// One wave = 16 nodes.  gi = aggr @ Wih^T + bih (K=128), gh = hs @ Whh^T + bhh (K=64).
// r/z gates use gi+gh sums -> shared accumulators; n-gate kept separate.
__global__ __launch_bounds__(256)
void gru_kernel(const float* __restrict__ aggr,       // [N, NDIM]
                float*  __restrict__ hsf,             // [N, HD] f32 (in-place)
                bf16_t* __restrict__ hsb,             // [N, HD] bf16 (in-place)
                const bf16_t* __restrict__ wih,       // [3*HD, NDIM] bf16
                const bf16_t* __restrict__ whh,       // [3*HD, HD] bf16 (global, cache-hot)
                const float* __restrict__ bih,        // [3*HD]
                const float* __restrict__ bhh)        // [3*HD]
{
  __shared__ bf16_t ldsWih[3 * HD * NDIM];            // 48 KB
  {
    const unsigned* s = (const unsigned*)wih;
    unsigned* d = (unsigned*)ldsWih;
    for (int i = threadIdx.x; i < 3 * HD * NDIM / 2; i += 256) d[i] = s[i];
  }
  __syncthreads();

  const int lane  = threadIdx.x & 31;
  const int wave  = threadIdx.x >> 5;
  const int tile  = blockIdx.x * 8 + wave;
  const int nbase = tile * 16;
  if (nbase >= N_NODES) return;          // wave-uniform guard

  const int mcol  = lane & 15;
  const int khalf = lane >> 4;
  const int lo    = khalf * 16;

  const int node = nbase + mcol;
  const float*  xr = aggr + (long long)node * NDIM;
  const bf16_t* hr = hsb  + (long long)node * HD;

  v8f accRZ[8];   // cols 0..127  (r, z) = gi + gh
  v8f accIN[4];   // cols 128..191 i_n
  v8f accHN[4];   // cols 128..191 h_n
#pragma unroll
  for (int t = 0; t < 8; ++t)
#pragma unroll
    for (int r = 0; r < 8; ++r) accRZ[t][r] = 0.0f;
#pragma unroll
  for (int t = 0; t < 4; ++t)
#pragma unroll
    for (int r = 0; r < 8; ++r) { accIN[t][r] = 0.0f; accHN[t][r] = 0.0f; }

  // gi: K = 128 over aggr (f32 -> bf16 on the fly)
#pragma unroll
  for (int s = 0; s < 4; ++s) {
    v16bf A;
#pragma unroll
    for (int i = 0; i < 8; ++i) {
      A[i]     = (bf16_t)xr[s * 32 + lo + i];
      A[8 + i] = (bf16_t)xr[s * 32 + lo + 8 + i];
    }
    const int k0 = s * 32 + khalf * 16;
#pragma unroll
    for (int t = 0; t < 8; ++t) {
      const int n = t * 16 + mcol;
      v16bf B = *(const v16bf*)(ldsWih + n * NDIM + k0);
      accRZ[t] = wmma_bf16(A, B, accRZ[t]);
    }
#pragma unroll
    for (int t = 0; t < 4; ++t) {
      const int n = (8 + t) * 16 + mcol;
      v16bf B = *(const v16bf*)(ldsWih + n * NDIM + k0);
      accIN[t] = wmma_bf16(A, B, accIN[t]);
    }
  }

  // gh: K = 64 over hidden state (already bf16)
#pragma unroll
  for (int s = 0; s < 2; ++s) {
    v8bf a0 = *(const v8bf*)(hr + s * 32 + lo);
    v8bf a1 = *(const v8bf*)(hr + s * 32 + lo + 8);
    v16bf A;
#pragma unroll
    for (int i = 0; i < 8; ++i) { A[i] = a0[i]; A[8 + i] = a1[i]; }
    const int k0 = s * 32 + khalf * 16;
#pragma unroll
    for (int t = 0; t < 8; ++t) {
      const int n = t * 16 + mcol;
      v16bf B = *(const v16bf*)(whh + n * HD + k0);
      accRZ[t] = wmma_bf16(A, B, accRZ[t]);
    }
#pragma unroll
    for (int t = 0; t < 4; ++t) {
      const int n = (8 + t) * 16 + mcol;
      v16bf B = *(const v16bf*)(whh + n * HD + k0);
      accHN[t] = wmma_bf16(A, B, accHN[t]);
    }
  }

  // gates — element-aligned across tiles in the 16x16 f32 C layout
#pragma unroll
  for (int t = 0; t < 4; ++t) {
    const int cR = t * 16 + mcol;        // column within HD
    const int cZ = HD + cR;
    const int cN = 2 * HD + cR;
    const float bR  = bih[cR] + bhh[cR];
    const float bZ  = bih[cZ] + bhh[cZ];
    const float biN = bih[cN];
    const float bhN = bhh[cN];
#pragma unroll
    for (int r = 0; r < 8; ++r) {
      const long long m = nbase + r + khalf * 8;
      const float rg = 1.0f / (1.0f + __expf(-(accRZ[t][r] + bR)));
      const float zg = 1.0f / (1.0f + __expf(-(accRZ[4 + t][r] + bZ)));
      const float ng = tanhf((accIN[t][r] + biN) + rg * (accHN[t][r] + bhN));
      const float hp = hsf[m * HD + cR];
      const float hn = (1.0f - zg) * ng + zg * hp;
      hsf[m * HD + cR] = hn;
      hsb[m * HD + cR] = (bf16_t)hn;
    }
  }
}

// ------------------------------------------------------------------ combine
__global__ void combine_kernel(const float* __restrict__ hsF,
                               const float* __restrict__ hsB,
                               float* __restrict__ out) {
  int i = blockIdx.x * blockDim.x + threadIdx.x;
  if (i >= N_NODES * HD) return;
  int n = i / HD, c = i % HD;
  out[(long long)n * NDIM + c]      = hsF[i];
  out[(long long)n * NDIM + HD + c] = hsB[i];
}

// ---------------------------------------------------------------------------
extern "C" void kernel_launch(void* const* d_in, const int* in_sizes, int n_in,
                              void* d_out, int out_size, void* d_ws, size_t ws_size,
                              hipStream_t stream) {
  const float*     h    = (const float*)d_in[0];
  const long long* ei   = (const long long*)d_in[1];   // int64 [2, E]
  const float* fw_msg_W = (const float*)d_in[2];
  const float* fw_msg_b = (const float*)d_in[3];
  const float* fw_Wih   = (const float*)d_in[4];
  const float* fw_Whh   = (const float*)d_in[5];
  const float* fw_bih   = (const float*)d_in[6];
  const float* fw_bhh   = (const float*)d_in[7];
  const float* bw_msg_W = (const float*)d_in[8];
  const float* bw_msg_b = (const float*)d_in[9];
  const float* bw_Wih   = (const float*)d_in[10];
  const float* bw_Whh   = (const float*)d_in[11];
  const float* bw_bih   = (const float*)d_in[12];
  const float* bw_bhh   = (const float*)d_in[13];

  // workspace carve-out (256-byte aligned slabs)
  char* ws = (char*)d_ws;
  size_t off = 0;
  auto carve = [&](size_t bytes) -> void* {
    void* p = ws + off;
    off += (bytes + 255) & ~(size_t)255;
    return p;
  };
  float*  hsF    = (float*)carve((size_t)N_NODES * HD * sizeof(float));
  float*  hsB    = (float*)carve((size_t)N_NODES * HD * sizeof(float));
  bf16_t* hsFb   = (bf16_t*)carve((size_t)N_NODES * HD * sizeof(bf16_t));
  bf16_t* hsBb   = (bf16_t*)carve((size_t)N_NODES * HD * sizeof(bf16_t));
  float*  aggr   = (float*)carve((size_t)N_NODES * NDIM * sizeof(float));
  bf16_t* wmsgF  = (bf16_t*)carve((size_t)NLAYERS * NDIM * NDIM * sizeof(bf16_t));
  bf16_t* wmsgB  = (bf16_t*)carve((size_t)NLAYERS * NDIM * NDIM * sizeof(bf16_t));
  bf16_t* wihF   = (bf16_t*)carve((size_t)NLAYERS * 3 * HD * NDIM * sizeof(bf16_t));
  bf16_t* wihB   = (bf16_t*)carve((size_t)NLAYERS * 3 * HD * NDIM * sizeof(bf16_t));
  bf16_t* whhF   = (bf16_t*)carve((size_t)NLAYERS * 3 * HD * HD * sizeof(bf16_t));
  bf16_t* whhB   = (bf16_t*)carve((size_t)NLAYERS * 3 * HD * HD * sizeof(bf16_t));

  const long long* srcA = ei;            // edge_index[0]
  const long long* dstA = ei + N_EDGES;  // edge_index[1]

  auto cgrid = [](int n) { return (n + 255) / 256; };

  // weight conversion (bf16)
  f32_to_bf16_kernel<<<cgrid(NLAYERS * NDIM * NDIM), 256, 0, stream>>>(fw_msg_W, wmsgF, NLAYERS * NDIM * NDIM);
  f32_to_bf16_kernel<<<cgrid(NLAYERS * NDIM * NDIM), 256, 0, stream>>>(bw_msg_W, wmsgB, NLAYERS * NDIM * NDIM);
  f32_to_bf16_kernel<<<cgrid(NLAYERS * 3 * HD * NDIM), 256, 0, stream>>>(fw_Wih, wihF, NLAYERS * 3 * HD * NDIM);
  f32_to_bf16_kernel<<<cgrid(NLAYERS * 3 * HD * NDIM), 256, 0, stream>>>(bw_Wih, wihB, NLAYERS * 3 * HD * NDIM);
  f32_to_bf16_kernel<<<cgrid(NLAYERS * 3 * HD * HD), 256, 0, stream>>>(fw_Whh, whhF, NLAYERS * 3 * HD * HD);
  f32_to_bf16_kernel<<<cgrid(NLAYERS * 3 * HD * HD), 256, 0, stream>>>(bw_Whh, whhB, NLAYERS * 3 * HD * HD);

  prep_nodes_kernel<<<cgrid(N_NODES * HD), 256, 0, stream>>>(h, hsF, hsB, hsFb, hsBb);

  const int msgBlocks = (N_EDGES / 16 + 7) / 8;   // 6250
  const int gruBlocks = ((N_NODES + 15) / 16 + 7) / 8;

  // forward direction: gather (src, dst), scatter to dst
  for (int l = 0; l < NLAYERS; ++l) {
    zero_f32_kernel<<<cgrid(N_NODES * NDIM), 256, 0, stream>>>(aggr, N_NODES * NDIM);
    msg_kernel<<<msgBlocks, 256, 0, stream>>>(hsFb, srcA, dstA,
                                              wmsgF + (size_t)l * NDIM * NDIM,
                                              fw_msg_b + (size_t)l * NDIM, aggr);
    gru_kernel<<<gruBlocks, 256, 0, stream>>>(aggr, hsF, hsFb,
                                              wihF + (size_t)l * 3 * HD * NDIM,
                                              whhF + (size_t)l * 3 * HD * HD,
                                              fw_bih + (size_t)l * 3 * HD,
                                              fw_bhh + (size_t)l * 3 * HD);
  }
  // backward direction: gather (dst, src), scatter to src
  for (int l = 0; l < NLAYERS; ++l) {
    zero_f32_kernel<<<cgrid(N_NODES * NDIM), 256, 0, stream>>>(aggr, N_NODES * NDIM);
    msg_kernel<<<msgBlocks, 256, 0, stream>>>(hsBb, dstA, srcA,
                                              wmsgB + (size_t)l * NDIM * NDIM,
                                              bw_msg_b + (size_t)l * NDIM, aggr);
    gru_kernel<<<gruBlocks, 256, 0, stream>>>(aggr, hsB, hsBb,
                                              wihB + (size_t)l * 3 * HD * NDIM,
                                              whhB + (size_t)l * 3 * HD * HD,
                                              bw_bih + (size_t)l * 3 * HD,
                                              bw_bhh + (size_t)l * 3 * HD);
  }

  combine_kernel<<<cgrid(N_NODES * HD), 256, 0, stream>>>(hsF, hsB, (float*)d_out);
}
